// TypeGraphModel_52183852646539
// MI455X (gfx1250) — matrix operand order repaired and checked by reading, not compiled, not run on hardware
//
#include <hip/hip_runtime.h>
#include <hip/hip_bf16.h>
#include <hip/hip_fp16.h>

typedef __attribute__((ext_vector_type(16))) _Float16 v16h;
typedef __attribute__((ext_vector_type(8)))  _Float16 v8h;
typedef __attribute__((ext_vector_type(8)))  float    v8f;

#define NS   1024   // sensors
#define DSIG 1024   // signal length
#define EMB  128
#define CCH  64
#define TPP  16
#define NHE  4
#define HDD  32
#define NLAY 2
#define FFD  2048
#define H1W  1028   // padded position rows (2 halo each side)

__device__ __forceinline__ float gelu_exact(float v){
  return 0.5f * v * (1.f + erff(v * 0.70710678118654752f));
}

__device__ __forceinline__ v16h frag_join(v8h lo, v8h hi){
  v16h r;
  #pragma unroll
  for (int i=0;i<8;i++){ r[i]=lo[i]; r[i+8]=hi[i]; }
  return r;
}
// A fragment (16x32 f16): element i -> K = laneHi*8 + (i&7) + (i>=8?16:0)
__device__ __forceinline__ v16h load_a(const _Float16* base, int laneHi){
  const v8h* p = (const v8h*)(base + (laneHi<<3));
  return frag_join(p[0], p[2]);
}
// B fragment (32x16 f16, weight/NK layout): element i -> K = laneHi*8 + (i&7) + (i>=8?16:0)
__device__ __forceinline__ v16h load_b(const _Float16* base, int laneHi){
  const v8h* p = (const v8h*)(base + (laneHi<<3));
  return frag_join(p[0], p[2]);
}

// ---- conv1 + BN + GELU -> h1t (N, 1028, 32) f16, position-major, zero halo ----
__global__ void k_conv1(const float* __restrict__ x, const float* __restrict__ w,
                        const float* __restrict__ b, const float* __restrict__ g,
                        const float* __restrict__ bb, const float* __restrict__ m,
                        const float* __restrict__ vv, _Float16* __restrict__ h1t){
  int s = blockIdx.x;
  __shared__ float ws[32*7];
  __shared__ float sc[32], sh[32];
  int tid = threadIdx.x;
  if (tid < 224) ws[tid] = w[tid];
  if (tid < 32){
    float scale = g[tid] * rsqrtf(vv[tid] + 1e-5f);
    sc[tid] = scale;
    sh[tid] = bb[tid] - m[tid]*scale;
  }
  _Float16* hs = h1t + (size_t)s*H1W*32;
  if (tid < 64){ hs[tid] = (_Float16)0.f; hs[1026*32 + tid] = (_Float16)0.f; }
  __syncthreads();
  const float* xs = x + (size_t)s*DSIG;
  for (int p = tid; p < DSIG; p += blockDim.x){
    float xv[7];
    #pragma unroll
    for (int t=0;t<7;t++){ int q=p+t-3; xv[t] = (q>=0 && q<DSIG)? xs[q] : 0.f; }
    _Float16 row[32];
    #pragma unroll 4
    for (int c=0;c<32;c++){
      float acc = b[c];
      #pragma unroll
      for (int t=0;t<7;t++) acc += ws[c*7+t]*xv[t];
      acc = acc*sc[c] + sh[c];
      row[c] = (_Float16)gelu_exact(acc);
    }
    _Float16* dst = hs + (size_t)(p+2)*32;
    #pragma unroll
    for (int c=0;c<32;c++) dst[c] = row[c];
  }
}

// ---- f32 -> f16 convert ----
__global__ void k_f2h(const float* __restrict__ src, _Float16* __restrict__ dst, int n){
  int i = blockIdx.x*blockDim.x + threadIdx.x;
  if (i < n) dst[i] = (_Float16)src[i];
}

// ---- rearrange conv2 weights: w2r[t][ch][c] = conv2_w[ch][c][t]  (5,64,32) f16 ----
__global__ void k_w2r(const float* __restrict__ w, _Float16* __restrict__ o){
  int idx = blockIdx.x*blockDim.x + threadIdx.x;
  if (idx >= 5*64*32) return;
  int t = idx >> 11;
  int r = idx & 2047;
  int ch = r >> 5, c = r & 31;
  o[idx] = (_Float16)w[ch*160 + c*5 + t];
}

// ---- V^T per head: vt[h][d][n] = qkvh[n][2*EMB + h*HDD + d]  (4,32,1024) f16 ----
__global__ void k_vt(const _Float16* __restrict__ qkvh, _Float16* __restrict__ vt){
  int idx = blockIdx.x*blockDim.x + threadIdx.x;
  if (idx >= NHE*HDD*NS) return;
  int n = idx & (NS-1);
  int d = (idx >> 10) & (HDD-1);
  int h = idx >> 15;
  vt[idx] = qkvh[(size_t)n*3*EMB + 2*EMB + h*HDD + d];
}

// ---- conv2 as 5 shifted K=32 WMMA GEMMs + BN + GELU + avg pool -> h2 (N, 64*16) ----
__global__ void k_conv2pool(const _Float16* __restrict__ h1t,
                            const _Float16* __restrict__ w2r,
                            const float* __restrict__ cb,
                            const float* __restrict__ g, const float* __restrict__ bb,
                            const float* __restrict__ m, const float* __restrict__ vv,
                            float* __restrict__ h2, _Float16* __restrict__ h2h){
  int s = blockIdx.x;
  __shared__ float pool[CCH*TPP];
  int tid = threadIdx.x;
  for (int i=tid;i<CCH*TPP;i+=blockDim.x) pool[i]=0.f;
  __syncthreads();
  int wave = tid>>5, lane = tid&31, laneLo = lane&15, laneHi = (lane>>4)&1;
  const _Float16* hs = h1t + (size_t)s*H1W*32;
  // 256 tile-jobs: (nt position-tile 0..63) x (mt channel-tile 0..3)
  for (int job = wave; job < 256; job += 8){
    int nt = job >> 2, mt = job & 3;
    int prow = nt*16 + laneLo;            // padded row index (pos - 2 + 2)
    int ch   = mt*16 + laneLo;
    v8f acc = (v8f){0.f,0.f,0.f,0.f,0.f,0.f,0.f,0.f};
    #pragma unroll
    for (int t=0;t<5;t++){
      v16h bf = load_b(hs + (size_t)(prow + t)*32, laneHi);      // h1t[prow+t][0..31]
      v16h af = load_a(w2r + t*2048 + ch*32, laneHi);            // w2r[t][ch][0..31]
      acc = __builtin_amdgcn_wmma_f32_16x16x32_f16(false, af, false, bf,
                                                   (short)0, acc, false, false);
    }
    int bin = nt >> 2;
    #pragma unroll
    for (int r=0;r<8;r++){
      int c = mt*16 + r + (laneHi<<3);
      float val = acc[r] + cb[c];
      float scale = g[c]*rsqrtf(vv[c]+1e-5f);
      val = (val - m[c])*scale + bb[c];
      val = gelu_exact(val);
      val += __shfl_xor(val,1,32);
      val += __shfl_xor(val,2,32);
      val += __shfl_xor(val,4,32);
      val += __shfl_xor(val,8,32);
      if (laneLo==0) atomicAdd(&pool[c*TPP+bin], val);
    }
  }
  __syncthreads();
  for (int i=tid;i<CCH*TPP;i+=blockDim.x){
    float o = pool[i]*(1.f/64.f);
    h2 [(size_t)s*CCH*TPP + i] = o;
    h2h[(size_t)s*CCH*TPP + i] = (_Float16)o;
  }
}

// ---- generic WMMA GEMM: Y = act(alpha * A(MxK) * B(NoutxK)^T + bias + res) ----
__global__ void k_gemm(const _Float16* __restrict__ A, int lda,
                       const _Float16* __restrict__ B, int ldb,
                       const float* __restrict__ bias,
                       const float* __restrict__ res, int ldres,
                       float* __restrict__ Y, int ldy,
                       _Float16* __restrict__ Yh, int ldyh,
                       int M, int Nout, int K, int act, float alpha){
  int tid = threadIdx.x;
  int wave = tid>>5, lane = tid&31, laneLo = lane&15, laneHi = (lane>>4)&1;
  int n0 = (blockIdx.y*8 + wave)*16;
  if (n0 >= Nout) return;
  int m0 = blockIdx.x*16;
  int arow = m0 + laneLo;
  int bcol = n0 + laneLo;
  bool colOK = bcol < Nout;
  const _Float16* Ap = A + (size_t)arow*lda;
  const _Float16* Bp = B + (size_t)(colOK ? bcol : n0)*ldb;  // clamp to safe row
  v8f acc = (v8f){0.f,0.f,0.f,0.f,0.f,0.f,0.f,0.f};
  for (int k0=0;k0<K;k0+=32){
    __builtin_prefetch(Ap + k0 + 256, 0, 3);
    v16h a = load_a(Ap + k0, laneHi);
    v16h b = load_b(Bp + k0, laneHi);
    if (!colOK){
      #pragma unroll
      for (int i=0;i<16;i++) b[i] = (_Float16)0.f;
    }
    acc = __builtin_amdgcn_wmma_f32_16x16x32_f16(false, a, false, b,
                                                 (short)0, acc, false, false);
  }
  #pragma unroll
  for (int r=0;r<8;r++){
    int row = m0 + r + (laneHi<<3);
    int col = n0 + laneLo;
    if (col < Nout){
      float v = alpha*acc[r];
      if (bias) v += bias[col];
      if (res)  v += res[(size_t)row*ldres + col];
      if (act==1) v = fmaxf(v, 0.f);
      if (Y)  Y [(size_t)row*ldy  + col] = v;
      if (Yh) Yh[(size_t)row*ldyh + col] = (_Float16)v;
    }
  }
}

// ---- row softmax: S(f32, rows x n) -> P(f16) ----
__global__ void k_softmax(const float* __restrict__ S, _Float16* __restrict__ P, int n){
  __shared__ float red[256];
  int row = blockIdx.x, tid = threadIdx.x;
  const float* sr = S + (size_t)row*n;
  float mx = -1e30f;
  for (int j=tid;j<n;j+=256) mx = fmaxf(mx, sr[j]);
  red[tid]=mx; __syncthreads();
  for (int s2=128;s2>0;s2>>=1){ if (tid<s2) red[tid]=fmaxf(red[tid],red[tid+s2]); __syncthreads(); }
  mx = red[0]; __syncthreads();
  float sum = 0.f;
  for (int j=tid;j<n;j+=256) sum += expf(sr[j]-mx);
  red[tid]=sum; __syncthreads();
  for (int s2=128;s2>0;s2>>=1){ if (tid<s2) red[tid]+=red[tid+s2]; __syncthreads(); }
  float inv = 1.f/red[0];
  for (int j=tid;j<n;j+=256) P[(size_t)row*n + j] = (_Float16)(expf(sr[j]-mx)*inv);
}

// ---- LayerNorm over EMB=128 ----
__global__ void k_ln(const float* __restrict__ X, const float* __restrict__ g,
                     const float* __restrict__ b, float* __restrict__ Yo,
                     _Float16* __restrict__ Yh){
  __shared__ float red[128];
  int row = blockIdx.x, tid = threadIdx.x;
  float v = X[(size_t)row*EMB + tid];
  red[tid]=v; __syncthreads();
  for (int s=64;s>0;s>>=1){ if (tid<s) red[tid]+=red[tid+s]; __syncthreads(); }
  float mu = red[0]*(1.f/EMB); __syncthreads();
  float d = v - mu;
  red[tid]=d*d; __syncthreads();
  for (int s=64;s>0;s>>=1){ if (tid<s) red[tid]+=red[tid+s]; __syncthreads(); }
  float var = red[0]*(1.f/EMB);
  float o = d*rsqrtf(var+1e-5f)*g[tid] + b[tid];
  Yo[(size_t)row*EMB + tid] = o;
  Yh[(size_t)row*EMB + tid] = (_Float16)o;
}

// ---- fused all-pairs edge MLP: out[i,j] = relu(hi[i]+hj[j]+e1b) . e2w + e2b ----
__global__ void k_edge(const float* __restrict__ hi, const float* __restrict__ hj,
                       const float* __restrict__ e1b, const float* __restrict__ e2w,
                       const float* __restrict__ e2b, float* __restrict__ out){
  __shared__ float his[16*EMB], hjs[16*EMB], ew[EMB];
  int tid = threadIdx.x;
  int i0 = blockIdx.x*16, j0 = blockIdx.y*16;
  for (int idx=tid; idx<16*EMB; idx+=256){
    int r = idx>>7, e = idx&127;
    his[idx] = hi[(size_t)(i0+r)*EMB + e] + e1b[e];
    hjs[idx] = hj[(size_t)(j0+r)*EMB + e];
  }
  if (tid < EMB) ew[tid] = e2w[tid];
  __syncthreads();
  int ii = tid>>4, jj = tid&15;
  const float* hr = &his[ii*EMB];
  const float* hc = &hjs[jj*EMB];
  float acc = e2b[0];
  #pragma unroll 8
  for (int e=0;e<EMB;e++) acc += fmaxf(hr[e]+hc[e], 0.f)*ew[e];
  out[(size_t)(i0+ii)*NS + (j0+jj)] = acc;
}

// ======================================================================
extern "C" void kernel_launch(void* const* d_in, const int* in_sizes, int n_in,
                              void* d_out, int out_size, void* d_ws, size_t ws_size,
                              hipStream_t stream) {
  const float* x        = (const float*)d_in[0];
  const float* conv1_w  = (const float*)d_in[1];
  const float* conv1_b  = (const float*)d_in[2];
  const float* bn1_g = (const float*)d_in[3], *bn1_b = (const float*)d_in[4];
  const float* bn1_m = (const float*)d_in[5], *bn1_v = (const float*)d_in[6];
  const float* conv2_w  = (const float*)d_in[7];
  const float* conv2_b  = (const float*)d_in[8];
  const float* bn2_g = (const float*)d_in[9], *bn2_b = (const float*)d_in[10];
  const float* bn2_m = (const float*)d_in[11], *bn2_v = (const float*)d_in[12];
  const float* proj_w = (const float*)d_in[13], *proj_b = (const float*)d_in[14];
  const float* attn_in_w = (const float*)d_in[15], *attn_in_b = (const float*)d_in[16];
  const float* attn_out_w= (const float*)d_in[17], *attn_out_b= (const float*)d_in[18];
  const float* ln1_g = (const float*)d_in[19], *ln1_b = (const float*)d_in[20];
  const float* ff1_w = (const float*)d_in[21], *ff1_b = (const float*)d_in[22];
  const float* ff2_w = (const float*)d_in[23], *ff2_b = (const float*)d_in[24];
  const float* ln2_g = (const float*)d_in[25], *ln2_b = (const float*)d_in[26];
  const float* nh1_w = (const float*)d_in[27], *nh1_b = (const float*)d_in[28];
  const float* nh2_w = (const float*)d_in[29], *nh2_b = (const float*)d_in[30];
  const float* e1_w  = (const float*)d_in[31], *e1_b  = (const float*)d_in[32];
  const float* e2_w  = (const float*)d_in[33], *e2_b  = (const float*)d_in[34];
  const float* p1_w  = (const float*)d_in[35], *p1_b  = (const float*)d_in[36];
  const float* p2_w  = (const float*)d_in[37], *p2_b  = (const float*)d_in[38];

  float* out      = (float*)d_out;
  float* nodeOut  = out;                       // (N,3)
  float* edgeOut  = out + 3072;                // (N,N)
  float* projOut  = out + 3072 + NS*NS;        // (N,EMB)

  char* wsp = (char*)d_ws;
  size_t off = 0;
  auto alloc = [&](size_t bytes)->void* {
    void* p = (void*)(wsp + off);
    off += (bytes + 255) & ~(size_t)255;
    return p;
  };
  _Float16* h1t    = (_Float16*)alloc((size_t)NS*H1W*32*2);         // ~64 MiB
  float*    h2     = (float*)   alloc((size_t)NS*CCH*TPP*4);
  _Float16* h2h    = (_Float16*)alloc((size_t)NS*CCH*TPP*2);
  float*    z      = (float*)   alloc((size_t)NS*EMB*4);
  _Float16* zh     = (_Float16*)alloc((size_t)NS*EMB*2);
  float*    t      = (float*)   alloc((size_t)NS*EMB*4);
  _Float16* th     = (_Float16*)alloc((size_t)NS*EMB*2);
  float*    tmp    = (float*)   alloc((size_t)NS*EMB*4);
  _Float16* qkvh   = (_Float16*)alloc((size_t)NS*3*EMB*2);
  _Float16* vth    = (_Float16*)alloc((size_t)NHE*HDD*NS*2);
  float*    S      = (float*)   alloc((size_t)NHE*NS*NS*4);         // 16 MiB
  _Float16* Ph     = (_Float16*)alloc((size_t)NHE*NS*NS*2);         // 8 MiB
  _Float16* oh     = (_Float16*)alloc((size_t)NS*EMB*2);
  _Float16* f1h    = (_Float16*)alloc((size_t)NS*FFD*2);
  _Float16* n1h    = (_Float16*)alloc((size_t)NS*EMB*2);
  float*    hiB    = (float*)   alloc((size_t)NS*EMB*4);
  float*    hjB    = (float*)   alloc((size_t)NS*EMB*4);
  _Float16* p1h    = (_Float16*)alloc((size_t)NS*EMB*2);
  _Float16* w2r    = (_Float16*)alloc(5*64*32*2);
  _Float16* projwh = (_Float16*)alloc((size_t)EMB*CCH*TPP*2);
  _Float16* aiwh   = (_Float16*)alloc((size_t)NLAY*3*EMB*EMB*2);
  _Float16* aowh   = (_Float16*)alloc((size_t)NLAY*EMB*EMB*2);
  _Float16* f1wh   = (_Float16*)alloc((size_t)NLAY*FFD*EMB*2);
  _Float16* f2wh   = (_Float16*)alloc((size_t)NLAY*EMB*FFD*2);
  _Float16* nh1wh  = (_Float16*)alloc((size_t)EMB*EMB*2);
  _Float16* nh2wh  = (_Float16*)alloc((size_t)16*EMB*2);   // padded rows for safety
  _Float16* e1wh   = (_Float16*)alloc((size_t)EMB*2*EMB*2);
  _Float16* p1wh   = (_Float16*)alloc((size_t)EMB*EMB*2);
  _Float16* p2wh   = (_Float16*)alloc((size_t)EMB*EMB*2);

  auto cvt = [&](const float* src, _Float16* dst, int n){
    k_f2h<<<(n+255)/256, 256, 0, stream>>>(src, dst, n);
  };
  k_w2r<<<(5*64*32+255)/256, 256, 0, stream>>>(conv2_w, w2r);
  cvt(proj_w, projwh, EMB*CCH*TPP);
  cvt(attn_in_w, aiwh, NLAY*3*EMB*EMB);
  cvt(attn_out_w, aowh, NLAY*EMB*EMB);
  cvt(ff1_w, f1wh, NLAY*FFD*EMB);
  cvt(ff2_w, f2wh, NLAY*EMB*FFD);
  cvt(nh1_w, nh1wh, EMB*EMB);
  cvt(nh2_w, nh2wh, 3*EMB);
  cvt(e1_w, e1wh, EMB*2*EMB);
  cvt(p1_w, p1wh, EMB*EMB);
  cvt(p2_w, p2wh, EMB*EMB);

  auto gemm = [&](const _Float16* A, int lda, const _Float16* B, int ldb,
                  const float* bias, const float* res, int ldres,
                  float* Y, int ldy, _Float16* Yh, int ldyh,
                  int M, int Nout, int K, int act, float alpha){
    int ntiles = (Nout + 15)/16;
    dim3 grid(M/16, (ntiles + 7)/8);
    k_gemm<<<grid, 256, 0, stream>>>(A, lda, B, ldb, bias, res, ldres,
                                     Y, ldy, Yh, ldyh, M, Nout, K, act, alpha);
  };

  // ---- encoder ----
  k_conv1<<<NS, 256, 0, stream>>>(x, conv1_w, conv1_b, bn1_g, bn1_b, bn1_m, bn1_v, h1t);
  k_conv2pool<<<NS, 256, 0, stream>>>(h1t, w2r, conv2_b, bn2_g, bn2_b, bn2_m, bn2_v, h2, h2h);
  gemm(h2h, CCH*TPP, projwh, CCH*TPP, proj_b, nullptr, 0,
       z, EMB, zh, EMB, NS, EMB, CCH*TPP, 0, 1.f);

  // ---- transformer ----
  const float* tcur = z; const _Float16* tcurh = zh;
  const float scale = 0.17677669529663689f;   // 1/sqrt(32)
  for (int l = 0; l < NLAY; l++){
    gemm(tcurh, EMB, aiwh + (size_t)l*3*EMB*EMB, EMB, attn_in_b + l*3*EMB,
         nullptr, 0, nullptr, 0, qkvh, 3*EMB, NS, 3*EMB, EMB, 0, 1.f);
    k_vt<<<(NHE*HDD*NS+255)/256, 256, 0, stream>>>(qkvh, vth);
    for (int h = 0; h < NHE; h++){
      gemm(qkvh + h*HDD, 3*EMB, qkvh + EMB + h*HDD, 3*EMB, nullptr, nullptr, 0,
           S + (size_t)h*NS*NS, NS, nullptr, 0, NS, NS, HDD, 0, scale);
    }
    k_softmax<<<NHE*NS, 256, 0, stream>>>(S, Ph, NS);
    for (int h = 0; h < NHE; h++){
      gemm(Ph + (size_t)h*NS*NS, NS, vth + (size_t)h*HDD*NS, NS, nullptr, nullptr, 0,
           nullptr, 0, oh + h*HDD, EMB, NS, HDD, NS, 0, 1.f);
    }
    gemm(oh, EMB, aowh + (size_t)l*EMB*EMB, EMB, attn_out_b + l*EMB,
         tcur, EMB, tmp, EMB, nullptr, 0, NS, EMB, EMB, 0, 1.f);
    k_ln<<<NS, 128, 0, stream>>>(tmp, ln1_g + l*EMB, ln1_b + l*EMB, t, th);
    gemm(th, EMB, f1wh + (size_t)l*FFD*EMB, EMB, ff1_b + l*FFD,
         nullptr, 0, nullptr, 0, f1h, FFD, NS, FFD, EMB, 1, 1.f);
    gemm(f1h, FFD, f2wh + (size_t)l*EMB*FFD, FFD, ff2_b + l*EMB,
         t, EMB, tmp, EMB, nullptr, 0, NS, EMB, FFD, 0, 1.f);
    k_ln<<<NS, 128, 0, stream>>>(tmp, ln2_g + l*EMB, ln2_b + l*EMB, t, th);
    tcur = t; tcurh = th;
  }

  // ---- node head ----
  gemm(th, EMB, nh1wh, EMB, nh1_b, nullptr, 0, nullptr, 0, n1h, EMB,
       NS, EMB, EMB, 1, 1.f);
  gemm(n1h, EMB, nh2wh, EMB, nh2_b, nullptr, 0, nodeOut, 3, nullptr, 0,
       NS, 3, EMB, 0, 1.f);

  // ---- edge head ----
  gemm(th, EMB, e1wh,       2*EMB, nullptr, nullptr, 0, hiB, EMB, nullptr, 0,
       NS, EMB, EMB, 0, 1.f);
  gemm(th, EMB, e1wh + EMB, 2*EMB, nullptr, nullptr, 0, hjB, EMB, nullptr, 0,
       NS, EMB, EMB, 0, 1.f);
  {
    dim3 grid(NS/16, NS/16);
    k_edge<<<grid, 256, 0, stream>>>(hiB, hjB, e1_b, e2_w, e2_b, edgeOut);
  }

  // ---- projector (on z) ----
  gemm(zh, EMB, p1wh, EMB, p1_b, nullptr, 0, nullptr, 0, p1h, EMB,
       NS, EMB, EMB, 1, 1.f);
  gemm(p1h, EMB, p2wh, EMB, p2_b, nullptr, 0, projOut, EMB, nullptr, 0,
       NS, EMB, EMB, 0, 1.f);
}
